// FSQEncoder_28149215658038
// MI455X (gfx1250) — matrix operand order repaired and checked by reading, not compiled, not run on hardware
//
#include <hip/hip_runtime.h>
#include <hip/hip_bf16.h>
#include <math.h>

typedef __attribute__((ext_vector_type(16))) _Float16 v16h;
typedef __attribute__((ext_vector_type(8)))  _Float16 v8h;
typedef __attribute__((ext_vector_type(8)))  float    v8f;

#define kB   256
#define kT   2048
#define kW   64           // output columns per block
#define kCB  240          // codebook size 8*6*5

// Build a 16-element f16 fragment from two 16B-aligned LDS runs (2x ds_load_b128)
__device__ __forceinline__ v16h ldfrag(const _Float16* p0, const _Float16* p1) {
    v8h lo = *(const v8h*)p0;
    v8h hi = *(const v8h*)p1;
    v16h r;
    #pragma unroll
    for (int j = 0; j < 8; ++j) { r[j] = lo[j]; r[j + 8] = hi[j]; }
    return r;
}

// -------------------------------------------------------------------------
__global__ void fsq_zero_counts(float* __restrict__ counts) {
    if (threadIdx.x < kCB) counts[threadIdx.x] = 0.0f;
}

// -------------------------------------------------------------------------
// fused: conv1+relu -> conv2+relu -> 1x1 conv3 -> tanh/FSQ quantize ->
//        codes + LDS histogram -> 1x1 projection to 64 ch
__global__ void __launch_bounds__(256)
fsq_fused(const float* __restrict__ x,
          const float* __restrict__ w1, const float* __restrict__ b1,
          const float* __restrict__ w2, const float* __restrict__ b2,
          const float* __restrict__ w3, const float* __restrict__ b3,
          const float* __restrict__ wp, const float* __restrict__ bp,
          float* __restrict__ out, float* __restrict__ counts) {

    const int tb   = blockIdx.x * kW;   // tile base column in T
    const int b    = blockIdx.y;        // batch
    const int tid  = threadIdx.x;
    const int lane = tid & 31;
    const int wv   = tid >> 5;          // wave id, 8 waves
    const int hh   = lane >> 4;         // half-wave select
    const int nl   = lane & 15;         // 0..15

    // ---- LDS (~57.7 KB) ----------------------------------------------
    __shared__ __align__(16) _Float16 xsh[18 * 82];   // x tile (f16), cols tb-9..tb+72
    __shared__ __align__(16) _Float16 P1[80 * 64];    // im2col panel conv1 [col][K=64]
    __shared__ __align__(16) _Float16 h1s[32 * 80];   // h1, cols tb-8..tb+71
    __shared__ __align__(16) _Float16 P2[64 * 96];    // im2col panel conv2 [col][K=96]
    __shared__ __align__(16) _Float16 h2s[64 * 64];   // h2, cols tb..tb+63
    __shared__ __align__(16) _Float16 Aw1[32 * 64];   // conv1 weights, K padded 54->64
    __shared__ __align__(16) _Float16 Aw2[64 * 96];   // conv2 weights, K=96
    __shared__ float b1s[32], b2s[64], b3s[3], bps[64];
    __shared__ float w3s[3 * 64], wps[64 * 3];
    __shared__ float histS[kCB];
    __shared__ float qs[3 * kW];

    // ---- stage weights / biases / x tile ------------------------------
    for (int i = tid; i < 32 * 64; i += 256) {        // Aw1 row-major [M][K], zero-padded
        int m = i >> 6, k = i & 63;
        float v = 0.0f;
        if (k < 54) { int c = k / 3, kk = k - 3 * c; v = w1[(m * 18 + c) * 3 + kk]; }
        Aw1[i] = (_Float16)v;
    }
    for (int i = tid; i < 64 * 96; i += 256) {        // Aw2 row-major [M][K]
        int m = i / 96, k = i - 96 * m;
        int c = k / 3, kk = k - 3 * c;
        Aw2[i] = (_Float16)w2[(m * 32 + c) * 3 + kk];
    }
    if (tid < 32)  b1s[tid] = b1[tid];
    if (tid < 64)  b2s[tid] = b2[tid];
    if (tid < 3)   b3s[tid] = b3[tid];
    if (tid < 64)  bps[tid] = bp[tid];
    if (tid < 192) w3s[tid] = w3[tid];                // [3][64]
    if (tid < 192) wps[tid] = wp[tid];                // [64][3]
    if (tid < kCB) histS[tid] = 0.0f;
    for (int i = tid; i < 18 * 82; i += 256) {        // x tile, f16, zero-padded halo
        int c = i / 82, col = i - 82 * c;
        int g = tb - 9 + col;
        xsh[i] = (g >= 0 && g < kT) ? (_Float16)x[((size_t)b * 18 + c) * kT + g]
                                    : (_Float16)0.0f;
    }
    __syncthreads();

    // ---- build conv1 im2col panel P1[col 0..79][k 0..63] ---------------
    // h1 local col p = col-8; element k<54: c=k/3, kk=k%3 -> xsh[c][col+kk]
    for (int i = tid; i < 80 * 64; i += 256) {
        int col = i >> 6, k = i & 63;
        _Float16 v = (_Float16)0.0f;
        if (k < 54) { int c = k / 3, kk = k - 3 * c; v = xsh[c * 82 + col + kk]; }
        P1[i] = v;
    }
    __syncthreads();

    // ---- conv1 WMMA: h1[32][80] = relu(W1[32x64pad] * P1^T) ------------
    for (int tile = wv; tile < 10; tile += 8) {       // 2 Mtiles x 5 Ntiles
        int mt = tile / 5, nt = tile - 5 * mt;
        int m0 = mt * 16 + nl;
        int col = nt * 16 + nl;
        v8f acc = {0.f, 0.f, 0.f, 0.f, 0.f, 0.f, 0.f, 0.f};
        #pragma unroll
        for (int ks = 0; ks < 2; ++ks) {
            const int kb = ks * 32;
            v16h a = ldfrag(&Aw1[m0 * 64 + kb + 8 * hh],
                            &Aw1[m0 * 64 + kb + 16 + 8 * hh]);
            v16h bm = ldfrag(&P1[col * 64 + kb + 8 * hh],
                             &P1[col * 64 + kb + 16 + 8 * hh]);
            acc = __builtin_amdgcn_wmma_f32_16x16x32_f16(false, a, false, bm,
                                                         (short)0, acc, false, false);
        }
        #pragma unroll
        for (int r = 0; r < 8; ++r) {
            int m = mt * 16 + r + 8 * hh;
            h1s[m * 80 + col] = (_Float16)fmaxf(acc[r] + b1s[m], 0.0f);
        }
    }
    __syncthreads();

    // zero h1 columns outside [0,T): conv2 zero-pads h1, not conv1-of-padded-x
    for (int i = tid; i < 80; i += 256) {
        int g = tb - 8 + i;
        if (g < 0 || g >= kT)
            for (int m = 0; m < 32; ++m) h1s[m * 80 + i] = (_Float16)0.0f;
    }
    __syncthreads();

    // ---- build conv2 im2col panel P2[col 0..63][k 0..95] ---------------
    // k: c=k/3, kk=k%3 -> h1s[c][col+kk+7]   (h1 local p = col+kk-1, +8 offset)
    for (int i = tid; i < 64 * 96; i += 256) {
        int col = i / 96, k = i - 96 * col;
        int c = k / 3, kk = k - 3 * c;
        P2[i] = h1s[c * 80 + col + kk + 7];
    }
    __syncthreads();

    // ---- conv2 WMMA: h2[64][64] = relu(W2[64x96] * P2^T) ---------------
    for (int tile = wv; tile < 16; tile += 8) {       // 4 Mtiles x 4 Ntiles
        int mt = tile >> 2, nt = tile & 3;
        int m0 = mt * 16 + nl;
        int col = nt * 16 + nl;
        v8f acc = {0.f, 0.f, 0.f, 0.f, 0.f, 0.f, 0.f, 0.f};
        #pragma unroll
        for (int ks = 0; ks < 3; ++ks) {
            const int kb = ks * 32;
            v16h a = ldfrag(&Aw2[m0 * 96 + kb + 8 * hh],
                            &Aw2[m0 * 96 + kb + 16 + 8 * hh]);
            v16h bm = ldfrag(&P2[col * 96 + kb + 8 * hh],
                             &P2[col * 96 + kb + 16 + 8 * hh]);
            acc = __builtin_amdgcn_wmma_f32_16x16x32_f16(false, a, false, bm,
                                                         (short)0, acc, false, false);
        }
        #pragma unroll
        for (int r = 0; r < 8; ++r) {
            int m = mt * 16 + r + 8 * hh;
            h2s[m * 64 + col] = (_Float16)fmaxf(acc[r] + b2s[m], 0.0f);
        }
    }
    __syncthreads();

    // ---- conv3 (1x1, 3 ch) + tanh + FSQ quantize + codes + histogram ---
    if (tid < kW) {
        const int col = tid;
        float f0 = b3s[0], f1 = b3s[1], f2 = b3s[2];
        #pragma unroll 8
        for (int c = 0; c < 64; ++c) {
            float hv = (float)h2s[c * 64 + col];
            f0 += w3s[c] * hv;
            f1 += w3s[64 + c] * hv;
            f2 += w3s[128 + c] * hv;
        }
        const int   Ls[3] = {8, 6, 5};
        const float fv[3] = {f0, f1, f2};
        int idx[3];
        #pragma unroll
        for (int d = 0; d < 3; ++d) {
            float xbv = tanhf(fv[d]);
            float fi  = (xbv + 1.0f) * 0.5f * (float)(Ls[d] - 1);
            int   ii  = (int)floorf(fi + 0.5f);
            ii = ii < 0 ? 0 : (ii > Ls[d] - 1 ? Ls[d] - 1 : ii);
            idx[d] = ii;
            qs[d * kW + col] = -1.0f + 2.0f * (float)ii / (float)(Ls[d] - 1);
        }
        int code = idx[0] + idx[1] * 8 + idx[2] * 48;
        out[(size_t)b * kT + tb + col] = (float)code;
        atomicAdd(&histS[code], 1.0f);
    }
    __syncthreads();

    // flush histogram to global counts
    if (tid < kCB) {
        float hv = histS[tid];
        if (hv != 0.0f) atomicAdd(&counts[tid], hv);
    }

    // ---- projection: out64[64][kW] = Wp[64x3] * q + bp -----------------
    const size_t outOff = (size_t)kB * kT;
    for (int i = tid; i < 64 * kW; i += 256) {
        int m = i / kW, col = i - m * kW;
        float v = bps[m]
                + wps[m * 3 + 0] * qs[col]
                + wps[m * 3 + 1] * qs[kW + col]
                + wps[m * 3 + 2] * qs[2 * kW + col];
        out[outOff + ((size_t)b * 64 + m) * kT + tb + col] = v;
    }
}

// -------------------------------------------------------------------------
// probs, entropy, perplexity from global counts
__global__ void fsq_finalize(const float* __restrict__ counts, float* __restrict__ out) {
    __shared__ float red[256];
    const int tid = threadIdx.x;
    const size_t base = (size_t)kB * kT + (size_t)kB * 64 * kT; // perplexity slot
    float e = 0.0f;
    if (tid < kCB) {
        float p = counts[tid] * (1.0f / ((float)kB * (float)kT));
        out[base + 1 + tid] = p;                      // probs
        if (p > 0.0f) e = -p * logf(p);
    }
    red[tid] = e;
    __syncthreads();
    for (int s = 128; s > 0; s >>= 1) {
        if (tid < s) red[tid] += red[tid + s];
        __syncthreads();
    }
    if (tid == 0) out[base] = fmaxf(expf(red[0]), 1.0f);  // perplexity
}

// -------------------------------------------------------------------------
extern "C" void kernel_launch(void* const* d_in, const int* in_sizes, int n_in,
                              void* d_out, int out_size, void* d_ws, size_t ws_size,
                              hipStream_t stream) {
    const float* x  = (const float*)d_in[0];
    const float* w1 = (const float*)d_in[1];
    const float* b1 = (const float*)d_in[2];
    const float* w2 = (const float*)d_in[3];
    const float* b2 = (const float*)d_in[4];
    const float* w3 = (const float*)d_in[5];
    const float* b3 = (const float*)d_in[6];
    const float* wp = (const float*)d_in[7];
    const float* bp = (const float*)d_in[8];
    float* out    = (float*)d_out;
    float* counts = (float*)d_ws;   // 240 floats of scratch

    fsq_zero_counts<<<1, 256, 0, stream>>>(counts);

    dim3 grid(kT / kW, kB);         // 32 x 256 blocks
    fsq_fused<<<grid, 256, 0, stream>>>(x, w1, b1, w2, b2, w3, b3, wp, bp,
                                        out, counts);

    fsq_finalize<<<1, 256, 0, stream>>>(counts, out);
}
